// DialogueSNN_65120294142370
// MI455X (gfx1250) — matrix operand order repaired and checked by reading, not compiled, not run on hardware
//
#include <hip/hip_runtime.h>
#include <hip/hip_bf16.h>

// ---------------------------------------------------------------------------
// DialogueSNN forward for MI455X (gfx1250, wave32, WMMA)
//
// All GEMMs use v_wmma_f32_16x16x32_f16 (spikes/means are exact in f16,
// f32 accumulate). Sequential LIF recurrence runs as a persistent kernel
// (2 blocks x 16 waves) with membrane state held in WMMA accumulator
// fragments and spikes exchanged through LDS. The dominant output GEMM
// (4096x32000x256) uses 2x2 register blocking per wave to halve fragment
// traffic (L2-bound otherwise).
// ---------------------------------------------------------------------------

typedef _Float16 h8   __attribute__((ext_vector_type(8)));
typedef _Float16 v16h __attribute__((ext_vector_type(16)));
typedef float    v8f  __attribute__((ext_vector_type(8)));

#define BETA 0.85f
#define THR  1.0f

#define WMMA_F16(a, b, c) \
  __builtin_amdgcn_wmma_f32_16x16x32_f16(false, (a), false, (b), (short)0, (c), false, false)

// ---- fragment loaders (layouts per CDNA5 ISA 7.12.2) ----------------------
// A (16-bit, 16x32, row-major source [M][K]):
//   lane<16 : M=lane,    elems 0..7 -> K = 0..7,   elems 8..15 -> K = 16..23
//   lane>=16: M=lane-16, elems 0..7 -> K = 8..15,  elems 8..15 -> K = 24..31
__device__ inline v16h load_a_frag_g(const _Float16* __restrict__ A, int ldk,
                                     int mtile, int kk, int lane) {
  int m  = lane & 15;
  int kb = lane >> 4;
  const _Float16* base = A + (size_t)(mtile * 16 + m) * ldk + kk * 32 + kb * 8;
  h8 lo = *(const h8*)base;
  h8 hi = *(const h8*)(base + 16);
  return __builtin_shufflevector(lo, hi, 0,1,2,3,4,5,6,7,8,9,10,11,12,13,14,15);
}

// B (16-bit, 32x16) from BPACK layout: for k-block kk, column n, the 32 K
// values are contiguous:  Bp[((kk*N)+n)*32 + k]   -> two 16B loads per lane.
//   lane<16 : N=lane,    K=0..7,16..23 ; lane>=16: N=lane-16, K=8..15,24..31
__device__ inline v16h load_b_frag(const _Float16* __restrict__ Bp, int N,
                                   int ntile, int kk, int lane) {
  int n  = lane & 15;
  int kh = lane >> 4;
  const _Float16* base = Bp + ((size_t)kk * N + ntile * 16 + n) * 32 + kh * 8;
  h8 lo = *(const h8*)base;
  h8 hi = *(const h8*)(base + 16);
  return __builtin_shufflevector(lo, hi, 0,1,2,3,4,5,6,7,8,9,10,11,12,13,14,15);
}

// ---- BPACK packer: W is [N][K] row-major f32 (B[k][n] = W[n][k]) ----------
__global__ void snn_pack_b(const float* __restrict__ W, _Float16* __restrict__ out,
                           int N, int K) {
  size_t idx = (size_t)blockIdx.x * blockDim.x + threadIdx.x;
  if (idx >= (size_t)N * K) return;
  int n = (int)(idx / K), k = (int)(idx % K);
  out[(((size_t)(k >> 5) * N) + n) * 32 + (k & 31)] = (_Float16)W[idx];
}

// ---- gather embeddings: xemb[r=s*32+b][e] = embed[x[b,s]][e] (f16) --------
__global__ void snn_gather_emb(const int* __restrict__ x,
                               const float* __restrict__ embed,
                               _Float16* __restrict__ xemb) {
  int r = blockIdx.x;            // 0..4095
  int e = threadIdx.x;           // 0..127
  int s = r >> 5, b = r & 31;
  int tok = x[b * 128 + s];
  xemb[(size_t)r * 128 + e] = (_Float16)embed[(size_t)tok * 128 + e];
}

// ---- generic WMMA GEMM:  C = scale*(A@B) + bias ---------------------------
// A: AROW f16 [M][K]; Bp: BPACK f16 [K][N]; 8 waves/block, 1 tile/wave.
__global__ void __launch_bounds__(256)
snn_gemm(const _Float16* __restrict__ A,
         const _Float16* __restrict__ Bp,
         const float* __restrict__ bias, float scale,
         float* __restrict__ C, int M, int N, int K) {
  int lane  = threadIdx.x & 31;
  int wave  = threadIdx.x >> 5;
  int ntile = blockIdx.x * 8 + wave;
  int mtile = blockIdx.y;
  v8f acc = {};
  int kt = K >> 5;
  for (int kk = 0; kk < kt; ++kk) {
    v16h a = load_a_frag_g(A, K, mtile, kk, lane);
    v16h b = load_b_frag(Bp, N, ntile, kk, lane);
    acc = WMMA_F16(a, b, acc);
  }
  int n  = ntile * 16 + (lane & 15);
  float bv = bias ? bias[n] : 0.f;
  int mbase = mtile * 16 + ((lane >> 4) << 3);
#pragma unroll
  for (int i = 0; i < 8; ++i) {
    C[(size_t)(mbase + i) * N + n] = acc[i] * scale + bv;
  }
}

// ---- dominant output GEMM: pooled[4096,256] @ Wout^T -> out[B][S][V] ------
// 2x2 register blocking: wave computes a 32x32 macro-tile (4 WMMA chains per
// 4 fragment loads). Block = 8 waves as 2(M) x 4(N) -> 64 rows x 128 cols.
// Store permutes row r=(s*32+b) -> out[(b*128+s)*32000 + n].
__global__ void __launch_bounds__(256)
snn_gemm_out(const _Float16* __restrict__ A,   // pooled AROW [4096][256]
             const _Float16* __restrict__ Bp,  // BPACK K=256 N=32000
             const float* __restrict__ bias,   // bout[32000]
             float* __restrict__ C) {
  const int N = 32000, K = 256;
  int lane = threadIdx.x & 31;
  int wave = threadIdx.x >> 5;                     // 0..7
  int mt0 = blockIdx.y * 4 + (wave >> 2) * 2;      // m-tiles mt0, mt0+1
  int nt0 = blockIdx.x * 8 + (wave & 3) * 2;       // n-tiles nt0, nt0+1

  v8f a00 = {}, a01 = {}, a10 = {}, a11 = {};
#pragma unroll
  for (int kk = 0; kk < 8; ++kk) {
    v16h af0 = load_a_frag_g(A, K, mt0,     kk, lane);
    v16h af1 = load_a_frag_g(A, K, mt0 + 1, kk, lane);
    v16h bf0 = load_b_frag(Bp, N, nt0,     kk, lane);
    v16h bf1 = load_b_frag(Bp, N, nt0 + 1, kk, lane);
    a00 = WMMA_F16(af0, bf0, a00);
    a01 = WMMA_F16(af0, bf1, a01);
    a10 = WMMA_F16(af1, bf0, a10);
    a11 = WMMA_F16(af1, bf1, a11);
  }

  int nloc = lane & 15;
  int mloc0 = (lane >> 4) << 3;
  int n0 = nt0 * 16 + nloc, n1 = (nt0 + 1) * 16 + nloc;
  float bv0 = bias[n0], bv1 = bias[n1];
#pragma unroll
  for (int i = 0; i < 8; ++i) {
    int m0 = mt0 * 16 + mloc0 + i;
    int m1 = (mt0 + 1) * 16 + mloc0 + i;
    size_t r0 = ((size_t)(m0 & 31) * 128 + (m0 >> 5)) * N;   // (b*128+s)*N
    size_t r1 = ((size_t)(m1 & 31) * 128 + (m1 >> 5)) * N;
    C[r0 + n0] = a00[i] + bv0;
    C[r0 + n1] = a01[i] + bv1;
    C[r1 + n0] = a10[i] + bv0;
    C[r1 + n1] = a11[i] + bv1;
  }
}

// ---- persistent LIF recurrence (2 blocks x 512 thr = 16 waves) ------------
// Block owns batch rows [16*blockIdx.x, +16). Wave w owns output columns
// [16w,16w+16) of layers 2/3; mem2/mem3 live in WMMA D-fragment registers.
__global__ void __launch_bounds__(512)
snn_recurrence(const float* __restrict__ cur1,          // [4096][256] (s,b)
               const _Float16* __restrict__ w2p,        // BPACK K=256 N=256
               const _Float16* __restrict__ w3p,
               const float* __restrict__ b2v_, const float* __restrict__ b3v_,
               float* __restrict__ rec1, float* __restrict__ rec2,
               float* __restrict__ rec3,
               float* __restrict__ mem1o, float* __restrict__ mem2o,
               float* __restrict__ mem3o) {
  __shared__ _Float16 lds1[16 * 256];
  __shared__ _Float16 lds2[16 * 256];

  int tid  = threadIdx.x;
  int lane = tid & 31;
  int wave = tid >> 5;                 // 0..15: layer1 row AND layer2/3 ntile
  int wgBase = blockIdx.x * 16;

  // layer-1 scalar state: thread owns (row=wave, 8 cols starting at lane*8)
  int row  = wave;
  int col0 = lane * 8;
  int b    = wgBase + row;
  float m1[8];
#pragma unroll
  for (int j = 0; j < 8; ++j) m1[j] = 0.f;

  // layer-2/3 fragment state
  v8f m2 = {}, m3 = {};
  int nloc = lane & 15;
  int mloc0 = (lane >> 4) << 3;        // 0 or 8 (D-frag row base)
  float b2v = b2v_[wave * 16 + nloc];
  float b3v = b3v_[wave * 16 + nloc];
  int am = lane & 15, kb = lane >> 4;  // A-frag addressing

  float c1[8];
  for (int s = 0; s < 128; ++s) {
    const float* cp = cur1 + ((size_t)s * 32 + b) * 256 + col0;
#pragma unroll
    for (int j = 0; j < 8; ++j) c1[j] = cp[j];

    for (int t = 0; t < 20; ++t) {
      int step = s * 20 + t;
      // ---- layer 1 (no GEMM: current precomputed) ----
#pragma unroll
      for (int j = 0; j < 8; ++j) {
        float rst = m1[j] > THR ? 1.f : 0.f;
        m1[j] = BETA * m1[j] + c1[j] - rst;
        float spk = m1[j] > THR ? 1.f : 0.f;
        lds1[row * 256 + col0 + j] = (_Float16)spk;
        rec1[((size_t)step * 32 + b) * 256 + col0 + j] = spk;
      }
      __syncthreads();

      // ---- layer 2: mem2 = spk1@W2^T + (BETA*mem2 - reset + b2) ----
      {
        v8f acc;
#pragma unroll
        for (int i = 0; i < 8; ++i) {
          float rst = m2[i] > THR ? 1.f : 0.f;
          acc[i] = BETA * m2[i] - rst + b2v;
        }
#pragma unroll
        for (int kk = 0; kk < 8; ++kk) {
          h8 lo = *(const h8*)&lds1[am * 256 + kk * 32 + kb * 8];
          h8 hi = *(const h8*)&lds1[am * 256 + kk * 32 + 16 + kb * 8];
          v16h a = __builtin_shufflevector(lo, hi,
              0,1,2,3,4,5,6,7,8,9,10,11,12,13,14,15);
          v16h bb = load_b_frag(w2p, 256, wave, kk, lane);
          acc = WMMA_F16(a, bb, acc);
        }
        m2 = acc;
#pragma unroll
        for (int i = 0; i < 8; ++i) {
          float spk = m2[i] > THR ? 1.f : 0.f;
          int mr = mloc0 + i;
          lds2[mr * 256 + wave * 16 + nloc] = (_Float16)spk;
          rec2[((size_t)step * 32 + wgBase + mr) * 256 + wave * 16 + nloc] = spk;
        }
      }
      __syncthreads();

      // ---- layer 3 ----
      {
        v8f acc;
#pragma unroll
        for (int i = 0; i < 8; ++i) {
          float rst = m3[i] > THR ? 1.f : 0.f;
          acc[i] = BETA * m3[i] - rst + b3v;
        }
#pragma unroll
        for (int kk = 0; kk < 8; ++kk) {
          h8 lo = *(const h8*)&lds2[am * 256 + kk * 32 + kb * 8];
          h8 hi = *(const h8*)&lds2[am * 256 + kk * 32 + 16 + kb * 8];
          v16h a = __builtin_shufflevector(lo, hi,
              0,1,2,3,4,5,6,7,8,9,10,11,12,13,14,15);
          v16h bb = load_b_frag(w3p, 256, wave, kk, lane);
          acc = WMMA_F16(a, bb, acc);
        }
        m3 = acc;
#pragma unroll
        for (int i = 0; i < 8; ++i) {
          float spk = m3[i] > THR ? 1.f : 0.f;
          int mr = mloc0 + i;
          rec3[((size_t)step * 32 + wgBase + mr) * 256 + wave * 16 + nloc] = spk;
        }
      }
      __syncthreads();
    }
  }
  // final membrane states
#pragma unroll
  for (int j = 0; j < 8; ++j) mem1o[(size_t)b * 256 + col0 + j] = m1[j];
#pragma unroll
  for (int i = 0; i < 8; ++i) {
    int mr = mloc0 + i;
    mem2o[(size_t)(wgBase + mr) * 256 + wave * 16 + nloc] = m2[i];
    mem3o[(size_t)(wgBase + mr) * 256 + wave * 16 + nloc] = m3[i];
  }
}

// ---- pooled (mean over T of spk3), f16 AROW for the big GEMM --------------
__global__ void snn_pooled(const float* __restrict__ rec3,
                           _Float16* __restrict__ pooled) {
  int h = threadIdx.x;     // 256
  int r = blockIdx.x;      // 4096, r = s*32+b
  int s = r >> 5, b = r & 31;
  float acc = 0.f;
  for (int t = 0; t < 20; ++t)
    acc += rec3[(((size_t)(s * 20 + t)) * 32 + b) * 256 + h];
  pooled[(size_t)r * 256 + h] = (_Float16)(acc * 0.05f);
}

// ---- masked batch means for STDP (896 masked steps) -----------------------
__global__ void snn_masked_means(const float* __restrict__ rec1,
                                 const float* __restrict__ rec2,
                                 const float* __restrict__ rec3,
                                 _Float16* __restrict__ m1p,  // BPACK K=896 N=256
                                 _Float16* __restrict__ m2p,  // BPACK K=896 N=256
                                 _Float16* __restrict__ m2t,  // AROW [256][896]
                                 _Float16* __restrict__ m3t) {
  int h = threadIdx.x;     // 256
  int j = blockIdx.x;      // 896
  int n = (j / 7) * 20 + (j % 7) * 3;
  float s1 = 0.f, s2 = 0.f, s3 = 0.f;
  for (int b = 0; b < 32; ++b) {
    size_t base = ((size_t)n * 32 + b) * 256 + h;
    s1 += rec1[base]; s2 += rec2[base]; s3 += rec3[base];
  }
  const float inv = 1.f / 32.f;
  size_t bp = (((size_t)(j >> 5)) * 256 + h) * 32 + (j & 31);
  m1p[bp] = (_Float16)(s1 * inv);
  m2p[bp] = (_Float16)(s2 * inv);
  m2t[(size_t)h * 896 + j] = (_Float16)(s2 * inv);
  m3t[(size_t)h * 896 + j] = (_Float16)(s3 * inv);
}

// ---- P1^T[h][s] = sum over masked t of mean_b spk1 ------------------------
__global__ void snn_p1t(const float* __restrict__ rec1, _Float16* __restrict__ p1t) {
  int s = threadIdx.x;     // 128
  int h = blockIdx.x;      // 256
  float acc = 0.f;
  for (int tt = 0; tt < 7; ++tt) {
    int n = s * 20 + tt * 3;
    for (int b = 0; b < 32; ++b)
      acc += rec1[((size_t)n * 32 + b) * 256 + h];
  }
  p1t[(size_t)h * 128 + s] = (_Float16)(acc / 32.f);
}

// ---- e_mean[s][e] = mean_b embed[x[b,s]][e], BPACK (K=s, N=e) -------------
__global__ void snn_emean(const int* __restrict__ x, const float* __restrict__ embed,
                          _Float16* __restrict__ ep) {
  int e = threadIdx.x;     // 128
  int s = blockIdx.x;      // 128
  float acc = 0.f;
  for (int b = 0; b < 32; ++b)
    acc += embed[(size_t)x[b * 128 + s] * 128 + e];
  ep[(((size_t)(s >> 5)) * 128 + e) * 32 + (s & 31)] = (_Float16)(acc / 32.f);
}

// ---------------------------------------------------------------------------
extern "C" void kernel_launch(void* const* d_in, const int* in_sizes, int n_in,
                              void* d_out, int out_size, void* d_ws, size_t ws_size,
                              hipStream_t stream) {
  const int*   x     = (const int*)d_in[0];
  const float* embed = (const float*)d_in[1];
  const float* W1    = (const float*)d_in[2];
  const float* b1    = (const float*)d_in[3];
  const float* W2    = (const float*)d_in[4];
  const float* b2    = (const float*)d_in[5];
  const float* W3    = (const float*)d_in[6];
  const float* b3    = (const float*)d_in[7];
  const float* Wout  = (const float*)d_in[8];
  const float* bout  = (const float*)d_in[9];

  float* out  = (float*)d_out;                  // [32][128][32000]
  float* mem1 = out  + 131072000ull;            // [32][256]
  float* mem2 = mem1 + 8192;
  float* mem3 = mem2 + 8192;
  float* d1   = mem3 + 8192;                    // [256][128]
  float* d2   = d1   + 32768;                   // [256][256]
  float* d3   = d2   + 65536;
  float* rec1 = d3   + 65536;                   // [2560][32][256]
  float* rec2 = rec1 + 20971520ull;
  float* rec3 = rec2 + 20971520ull;

  // workspace partition (all sizes 256B-aligned)
  char* w = (char*)d_ws;
  float*     cur1   = (float*)w;     w += 4096ull * 256 * 4;   // [s*32+b][256]
  _Float16*  xemb   = (_Float16*)w;  w += 4096ull * 128 * 2;
  _Float16*  w1p    = (_Float16*)w;  w += 128ull * 256 * 2;
  _Float16*  w2p    = (_Float16*)w;  w += 256ull * 256 * 2;
  _Float16*  w3p    = (_Float16*)w;  w += 256ull * 256 * 2;
  _Float16*  woutp  = (_Float16*)w;  w += 256ull * 32000 * 2;
  _Float16*  pooled = (_Float16*)w;  w += 4096ull * 256 * 2;
  _Float16*  m1p    = (_Float16*)w;  w += 896ull * 256 * 2;
  _Float16*  m2p    = (_Float16*)w;  w += 896ull * 256 * 2;
  _Float16*  m2t    = (_Float16*)w;  w += 896ull * 256 * 2;
  _Float16*  m3t    = (_Float16*)w;  w += 896ull * 256 * 2;
  _Float16*  p1t    = (_Float16*)w;  w += 256ull * 128 * 2;
  _Float16*  ep     = (_Float16*)w;  w += 128ull * 128 * 2;

  // 1) pack weights to fragment-native BPACK f16
  snn_pack_b<<<(256 * 128 + 255) / 256, 256, 0, stream>>>(W1, w1p, 256, 128);
  snn_pack_b<<<(256 * 256 + 255) / 256, 256, 0, stream>>>(W2, w2p, 256, 256);
  snn_pack_b<<<(256 * 256 + 255) / 256, 256, 0, stream>>>(W3, w3p, 256, 256);
  snn_pack_b<<<(32000 * 256 + 255) / 256, 256, 0, stream>>>(Wout, woutp, 32000, 256);

  // 2) gather embeddings + per-token layer-1 currents
  snn_gather_emb<<<4096, 128, 0, stream>>>(x, embed, xemb);
  snn_gemm<<<dim3(2, 256), 256, 0, stream>>>(xemb, w1p, b1, 1.f, cur1,
                                             4096, 256, 128);

  // 3) sequential LIF recurrence (persistent, WMMA in-fragment state)
  snn_recurrence<<<2, 512, 0, stream>>>(cur1, w2p, w3p, b2, b3,
                                        rec1, rec2, rec3, mem1, mem2, mem3);

  // 4) pooled spikes + STDP traces
  snn_pooled<<<4096, 256, 0, stream>>>(rec3, pooled);
  snn_masked_means<<<896, 256, 0, stream>>>(rec1, rec2, rec3, m1p, m2p, m2t, m3t);
  snn_p1t<<<256, 128, 0, stream>>>(rec1, p1t);
  snn_emean<<<128, 128, 0, stream>>>(x, embed, ep);

  // 5) STDP GEMMs: d = (A_PRE*A_POST) * post^T @ pre
  const float c = 0.005f * (-0.003f);
  snn_gemm<<<dim3(1, 16), 256, 0, stream>>>(p1t, ep,  nullptr, c, d1, 256, 128, 128);
  snn_gemm<<<dim3(2, 16), 256, 0, stream>>>(m2t, m1p, nullptr, c, d2, 256, 256, 896);
  snn_gemm<<<dim3(2, 16), 256, 0, stream>>>(m3t, m2p, nullptr, c, d3, 256, 256, 896);

  // 6) output projection: 67 GFLOP, 524MB store floor (~23us @ 23.3TB/s);
  //    2x2-blocked waves keep fragment traffic ~4GB so HBM stores dominate.
  snn_gemm_out<<<dim3(250, 64), 256, 0, stream>>>(pooled, woutp, bout, out);
}
